// GraphAttentionLayer_19121194402239
// MI455X (gfx1250) — matrix operand order repaired and checked by reading, not compiled, not run on hardware
//
#include <hip/hip_runtime.h>
#include <hip/hip_bf16.h>

// ---------------------------------------------------------------------------
// Graph attention layer for MI455X (gfx1250, wave32, WMMA).
// B=4, N=1024, IN_F=OUT_F=512, E=32, H=8, HD=64.
// Memory-bound on edge_features (536 MB fp32): flash-attention structure
// reads edges exactly once, never materializes [B,H,N,N] scores in HBM.
// All matrix math through v_wmma_f32_16x16x32_f16 (f32 accumulate).
// ---------------------------------------------------------------------------

typedef __attribute__((ext_vector_type(16))) _Float16 v16h;
typedef __attribute__((ext_vector_type(8)))  float    v8f;

union HV {
    v16h v;
    _Float16 e[16];
    uint4 u4[2];
};

static __device__ __forceinline__ v8f wmma_f16(const v16h& a, const v16h& b, const v8f& c) {
    return __builtin_amdgcn_wmma_f32_16x16x32_f16(false, a, false, b, (short)0, c, false, false);
}

static __device__ __forceinline__ float halfmax16(float v) {
    // max across a 16-lane half (xor<16 never crosses halves on wave32)
    v = fmaxf(v, __shfl_xor(v, 1));
    v = fmaxf(v, __shfl_xor(v, 2));
    v = fmaxf(v, __shfl_xor(v, 4));
    v = fmaxf(v, __shfl_xor(v, 8));
    return v;
}
static __device__ __forceinline__ float halfsum16(float v) {
    v += __shfl_xor(v, 1);
    v += __shfl_xor(v, 2);
    v += __shfl_xor(v, 4);
    v += __shfl_xor(v, 8);
    return v;
}

#define BB   4
#define NN   1024
#define FF   512     // IN_F == OUT_F
#define EE   32
#define HH   8
#define HD   64

// ---------------------------------------------------------------------------
// fp32 -> fp16 conversion
// ---------------------------------------------------------------------------
__global__ void cvt_f16_kernel(const float* __restrict__ src, _Float16* __restrict__ dst, int n) {
    int i = blockIdx.x * blockDim.x + threadIdx.x;
    if (i < n) dst[i] = (_Float16)src[i];
}

// Build padded We^T operand: Weh[16][32] row h = We[h,:] (h<8), zeros above.
__global__ void build_we_kernel(const float* __restrict__ We, _Float16* __restrict__ Weh) {
    int i = threadIdx.x;                 // 512 threads
    int h = i >> 5, e = i & 31;
    Weh[i] = (_Float16)(h < HH ? We[h * EE + e] : 0.0f);
}

// ---------------------------------------------------------------------------
// Q/K/V projection: Y = X @ W^T + b. One 16x16 tile per wave, K=512 in 16
// WMMA steps. Q,K stored [B,H,N,HD] f16; V stored transposed [B,H,HD,N] f16
// so the attn*V B-operand loads are contiguous along K.
// ---------------------------------------------------------------------------
__global__ __launch_bounds__(256)
void qkv_gemm_kernel(const _Float16* __restrict__ Xh,
                     const _Float16* __restrict__ Wqh,
                     const _Float16* __restrict__ Wkh,
                     const _Float16* __restrict__ Wvh,
                     const float* __restrict__ bq,
                     const float* __restrict__ bk,
                     const float* __restrict__ bv,
                     _Float16* __restrict__ Qh,
                     _Float16* __restrict__ Kh,
                     _Float16* __restrict__ Vt) {
    const int wave = threadIdx.x >> 5;
    const int lane = threadIdx.x & 31;
    const int m16  = lane & 15;
    const int kg   = lane >> 4;
    const int which = blockIdx.z;                 // 0=Q 1=K 2=V
    const int nt = blockIdx.x * 8 + wave;         // 0..31  (output 16-col tile)
    const int mt = blockIdx.y;                    // 0..255 (row tile over B*N)
    const int m0 = mt * 16, o0 = nt * 16;

    const _Float16* W   = (which == 0) ? Wqh : (which == 1) ? Wkh : Wvh;
    const float*    bia = (which == 0) ? bq  : (which == 1) ? bk  : bv;

    v8f acc = {0.f, 0.f, 0.f, 0.f, 0.f, 0.f, 0.f, 0.f};
    for (int kk = 0; kk < FF; kk += 32) {
        HV a, b;
        // A: 16 rows of X, K-halves per wave32 f16 A layout
        const _Float16* ar = Xh + (size_t)(m0 + m16) * FF + kk;
        a.u4[0] = *(const uint4*)(ar + kg * 8);
        a.u4[1] = *(const uint4*)(ar + 16 + kg * 8);
        // B: columns are rows of W (Y = X W^T); lane holds col o0+m16,
        // K rows kk + kg*8 .. +7 and kk + 16 + kg*8 .. +7 (matches A layout)
        const _Float16* br = W + (size_t)(o0 + m16) * FF + kk + kg * 8;
        b.u4[0] = *(const uint4*)(br);
        b.u4[1] = *(const uint4*)(br + 16);
        acc = wmma_f16(a.v, b.v, acc);
    }

    const float bval = bia[o0 + m16];
    const int h  = o0 / HD;
    const int d0 = o0 % HD;
    if (which < 2) {
        _Float16* dst = (which == 0) ? Qh : Kh;
#pragma unroll
        for (int r = 0; r < 8; r++) {
            int row  = m0 + r + kg * 8;           // flat b*N+n
            int bIdx = row >> 10, nIdx = row & (NN - 1);
            dst[(((size_t)bIdx * HH + h) * NN + nIdx) * HD + d0 + m16] =
                (_Float16)(acc[r] + bval);
        }
    } else {
#pragma unroll
        for (int r = 0; r < 8; r++) {
            int row  = m0 + r + kg * 8;
            int bIdx = row >> 10, nIdx = row & (NN - 1);
            Vt[(((size_t)bIdx * HH + h) * HD + d0 + m16) * NN + nIdx] =
                (_Float16)(acc[r] + bval);
        }
    }
}

// ---------------------------------------------------------------------------
// Fused edge-bias + flash attention. Grid: (N/16, B); 256 threads = 8 waves,
// wave w == head w. Per 32-wide K tile:
//   Phase A: all waves cooperatively compute bias[qk(512), head(16)] via
//            WMMA (A = edge rows f32->f16, B = padded We^T), fold mask + be,
//            stage fp32 in LDS (stride 17 to dodge bank conflicts).
//   Phase B: per-head flash update: S = Q K^T (2x2 WMMA) * 1/sqrt(64) + bias,
//            online softmax (16-lane shfl reductions), P transposed through
//            LDS into WMMA A layout, O += P V (4 WMMA).
// Edge tensor is read exactly once across the whole dispatch.
// ---------------------------------------------------------------------------
__global__ __launch_bounds__(256)
void attn_kernel(const _Float16* __restrict__ Qh,
                 const _Float16* __restrict__ Kh,
                 const _Float16* __restrict__ Vt,
                 const float* __restrict__ edge,
                 const int* __restrict__ mask,
                 const _Float16* __restrict__ Weh,
                 const float* __restrict__ be,
                 _Float16* __restrict__ Oh) {
    __shared__ float    sBias[512 * 17];      // [qk row][head col], padded
    __shared__ _Float16 sP[8 * 16 * 40];      // per-wave P staging, 80B rows

    const int w    = threadIdx.x >> 5;        // wave id == head
    const int lane = threadIdx.x & 31;
    const int m16  = lane & 15;
    const int kg   = lane >> 4;
    const int b    = blockIdx.y;
    const int q0   = blockIdx.x * 16;

    // Loop-invariant B operand: We^T padded to 16 head-columns.
    HV weB;
    {
        const _Float16* p = Weh + m16 * EE + kg * 8;
        weB.u4[0] = *(const uint4*)p;
        weB.u4[1] = *(const uint4*)(p + 16);
    }
    const float beval = (m16 < HH) ? be[m16] : 0.0f;

    // Loop-invariant Q A-operands (d 0..31 and 32..63).
    HV qa0, qa1;
    {
        const _Float16* p = Qh + ((size_t)(b * HH + w) * NN + q0 + m16) * HD;
        qa0.u4[0] = *(const uint4*)(p + kg * 8);
        qa0.u4[1] = *(const uint4*)(p + 16 + kg * 8);
        qa1.u4[0] = *(const uint4*)(p + 32 + kg * 8);
        qa1.u4[1] = *(const uint4*)(p + 48 + kg * 8);
    }

    v8f Oacc[4];
    float mrow[8], lrow[8];
#pragma unroll
    for (int t = 0; t < 4; t++) Oacc[t] = (v8f){0.f, 0.f, 0.f, 0.f, 0.f, 0.f, 0.f, 0.f};
#pragma unroll
    for (int r = 0; r < 8; r++) { mrow[r] = -3.0e38f; lrow[r] = 0.0f; }

    const v8f zero8 = {0.f, 0.f, 0.f, 0.f, 0.f, 0.f, 0.f, 0.f};
    const size_t bh = (size_t)(b * HH + w);

    for (int k0 = 0; k0 < NN; k0 += 32) {
        // ---------------- Phase A: edge bias (cooperative) ----------------
#pragma unroll
        for (int t = 0; t < 4; t++) {
            const int base = w * 64 + t * 16;           // this wave's qk rows
            const int row  = base + m16;
            const int ql   = row >> 5, kl = row & 31;
            const float* ep = edge + (((size_t)b * NN + q0 + ql) * NN + k0 + kl) * EE;
            if (k0 + 32 < NN)
                __builtin_prefetch(ep + (size_t)32 * EE, 0, 0);  // next K tile
            union { float4 f4[4]; float f[16]; } ef;
            ef.f4[0] = *(const float4*)(ep + kg * 8);
            ef.f4[1] = *(const float4*)(ep + kg * 8 + 4);
            ef.f4[2] = *(const float4*)(ep + 16 + kg * 8);
            ef.f4[3] = *(const float4*)(ep + 16 + kg * 8 + 4);
            HV a;
#pragma unroll
            for (int i = 0; i < 16; i++) a.e[i] = (_Float16)ef.f[i];
            v8f c = wmma_f16(a.v, weB.v, zero8);
#pragma unroll
            for (int r = 0; r < 8; r++) {
                const int qk  = base + r + kg * 8;
                const int ql2 = qk >> 5, kl2 = qk & 31;
                const int mv  = mask[((size_t)b * NN + q0 + ql2) * NN + k0 + kl2];
                const float vv = c[r] + beval;
                sBias[qk * 17 + m16] = mv ? vv : -3.0e38f;
            }
        }
        __syncthreads();

        // ---------------- Phase B: flash attention (per head) -------------
        v8f S0 = zero8, S1 = zero8;
        {
            const _Float16* kp0 = Kh + (bh * NN + k0 + m16) * HD;        // k cols 0..15
            const _Float16* kp1 = Kh + (bh * NN + k0 + 16 + m16) * HD;   // k cols 16..31
            HV b00, b01, b10, b11;
            b00.u4[0] = *(const uint4*)(kp0 + kg * 8);
            b00.u4[1] = *(const uint4*)(kp0 + kg * 8 + 16);
            b01.u4[0] = *(const uint4*)(kp0 + 32 + kg * 8);
            b01.u4[1] = *(const uint4*)(kp0 + 32 + kg * 8 + 16);
            b10.u4[0] = *(const uint4*)(kp1 + kg * 8);
            b10.u4[1] = *(const uint4*)(kp1 + kg * 8 + 16);
            b11.u4[0] = *(const uint4*)(kp1 + 32 + kg * 8);
            b11.u4[1] = *(const uint4*)(kp1 + 32 + kg * 8 + 16);
            S0 = wmma_f16(qa0.v, b00.v, S0);
            S0 = wmma_f16(qa1.v, b01.v, S0);
            S1 = wmma_f16(qa0.v, b10.v, S1);
            S1 = wmma_f16(qa1.v, b11.v, S1);
        }

        float pv0[8], pv1[8], scl[8];
#pragma unroll
        for (int r = 0; r < 8; r++) {
            const int qrow = r + kg * 8;
            const float s0 = S0[r] * 0.125f + sBias[(qrow * 32 + m16) * 17 + w];
            const float s1 = S1[r] * 0.125f + sBias[(qrow * 32 + 16 + m16) * 17 + w];
            float mx = halfmax16(fmaxf(s0, s1));
            const float mnew = fmaxf(mrow[r], mx);
            const float p0 = __expf(s0 - mnew);
            const float p1 = __expf(s1 - mnew);
            const float rs = halfsum16(p0 + p1);
            const float sc = __expf(mrow[r] - mnew);
            lrow[r] = lrow[r] * sc + rs;
            mrow[r] = mnew;
            scl[r] = sc;
            pv0[r] = p0;
            pv1[r] = p1;
        }
#pragma unroll
        for (int t = 0; t < 4; t++)
#pragma unroll
            for (int r = 0; r < 8; r++) Oacc[t][r] *= scl[r];

        // Transpose P (C layout -> A layout) through per-wave LDS.
        _Float16* prow = &sP[w * 16 * 40];
#pragma unroll
        for (int r = 0; r < 8; r++) {
            const int qrow = r + kg * 8;
            prow[qrow * 40 + m16]      = (_Float16)pv0[r];
            prow[qrow * 40 + 16 + m16] = (_Float16)pv1[r];
        }
        asm volatile("s_wait_dscnt 0x0" ::: "memory");
        HV pa;
        {
            const uint4* pp = (const uint4*)(prow + (size_t)m16 * 40);
            pa.u4[0] = pp[kg];
            pa.u4[1] = pp[2 + kg];
        }

        // O += P @ V, B from transposed V (contiguous along K rows).
#pragma unroll
        for (int t = 0; t < 4; t++) {
            HV vb;
            const _Float16* vp = Vt + (bh * HD + t * 16 + m16) * NN + k0 + kg * 8;
            vb.u4[0] = *(const uint4*)vp;
            vb.u4[1] = *(const uint4*)(vp + 16);
            Oacc[t] = wmma_f16(pa.v, vb.v, Oacc[t]);
        }
        __syncthreads();   // protect sBias reuse next iteration
    }

    // Epilogue: O /= l, store f16 to [B, N, H*HD] for the output projection.
#pragma unroll
    for (int r = 0; r < 8; r++) {
        const float inv  = (lrow[r] > 0.0f) ? 1.0f / lrow[r] : 0.0f;
        const int   qrow = r + kg * 8;
        _Float16* op = Oh + ((size_t)b * NN + q0 + qrow) * FF + w * HD;
#pragma unroll
        for (int t = 0; t < 4; t++)
            op[t * 16 + m16] = (_Float16)(Oacc[t][r] * inv);
    }
}

// ---------------------------------------------------------------------------
// Output projection: out = O @ Wo^T + bo, fp32 result.
// ---------------------------------------------------------------------------
__global__ __launch_bounds__(256)
void out_gemm_kernel(const _Float16* __restrict__ Oh,
                     const _Float16* __restrict__ Woh,
                     const float* __restrict__ bo,
                     float* __restrict__ out) {
    const int wave = threadIdx.x >> 5;
    const int lane = threadIdx.x & 31;
    const int m16  = lane & 15;
    const int kg   = lane >> 4;
    const int nt = blockIdx.x * 8 + wave;
    const int mt = blockIdx.y;
    const int m0 = mt * 16, o0 = nt * 16;

    v8f acc = {0.f, 0.f, 0.f, 0.f, 0.f, 0.f, 0.f, 0.f};
    for (int kk = 0; kk < FF; kk += 32) {
        HV a, b;
        const _Float16* ar = Oh + (size_t)(m0 + m16) * FF + kk;
        a.u4[0] = *(const uint4*)(ar + kg * 8);
        a.u4[1] = *(const uint4*)(ar + 16 + kg * 8);
        const _Float16* br = Woh + (size_t)(o0 + m16) * FF + kk + kg * 8;
        b.u4[0] = *(const uint4*)(br);
        b.u4[1] = *(const uint4*)(br + 16);
        acc = wmma_f16(a.v, b.v, acc);
    }
    const float bval = bo[o0 + m16];
#pragma unroll
    for (int r = 0; r < 8; r++) {
        const int row = m0 + r + kg * 8;
        out[(size_t)row * FF + o0 + m16] = acc[r] + bval;
    }
}

// ---------------------------------------------------------------------------
extern "C" void kernel_launch(void* const* d_in, const int* in_sizes, int n_in,
                              void* d_out, int out_size, void* d_ws, size_t ws_size,
                              hipStream_t stream) {
    const float* node = (const float*)d_in[0];
    const float* edge = (const float*)d_in[1];
    const int*   mask = (const int*)d_in[2];
    const float* Wq = (const float*)d_in[3];
    const float* bq = (const float*)d_in[4];
    const float* Wk = (const float*)d_in[5];
    const float* bk = (const float*)d_in[6];
    const float* Wv = (const float*)d_in[7];
    const float* bv = (const float*)d_in[8];
    const float* We = (const float*)d_in[9];
    const float* be = (const float*)d_in[10];
    const float* Wo = (const float*)d_in[11];
    const float* bo = (const float*)d_in[12];
    float* out = (float*)d_out;

    // Workspace layout (f16 elements), ~23 MB total.
    _Float16* w = (_Float16*)d_ws;
    const size_t nBN  = (size_t)BB * NN;             // 4096
    const size_t nX   = nBN * FF;                    // 2M
    const size_t nW   = (size_t)FF * FF;             // 256K
    const size_t nQKV = (size_t)BB * HH * NN * HD;   // 2M
    _Float16* Xh  = w;                 size_t off = nX;
    _Float16* Wqh = w + off;           off += nW;
    _Float16* Wkh = w + off;           off += nW;
    _Float16* Wvh = w + off;           off += nW;
    _Float16* Woh = w + off;           off += nW;
    _Float16* Weh = w + off;           off += 16 * EE;
    _Float16* Qh  = w + off;           off += nQKV;
    _Float16* Kh  = w + off;           off += nQKV;
    _Float16* Vt  = w + off;           off += nQKV;
    _Float16* Oh  = w + off;           off += nX;

    // 1) f32 -> f16 conversions
    cvt_f16_kernel<<<(int)((nX + 255) / 256), 256, 0, stream>>>(node, Xh, (int)nX);
    cvt_f16_kernel<<<(int)((nW + 255) / 256), 256, 0, stream>>>(Wq, Wqh, (int)nW);
    cvt_f16_kernel<<<(int)((nW + 255) / 256), 256, 0, stream>>>(Wk, Wkh, (int)nW);
    cvt_f16_kernel<<<(int)((nW + 255) / 256), 256, 0, stream>>>(Wv, Wvh, (int)nW);
    cvt_f16_kernel<<<(int)((nW + 255) / 256), 256, 0, stream>>>(Wo, Woh, (int)nW);
    build_we_kernel<<<1, 512, 0, stream>>>(We, Weh);

    // 2) Q/K/V projections (WMMA)
    qkv_gemm_kernel<<<dim3(4, 256, 3), 256, 0, stream>>>(
        Xh, Wqh, Wkh, Wvh, bq, bk, bv, Qh, Kh, Vt);

    // 3) Fused edge-bias + flash attention (WMMA), edges read once
    attn_kernel<<<dim3(NN / 16, BB), 256, 0, stream>>>(
        Qh, Kh, Vt, edge, mask, Weh, be, Oh);

    // 4) Output projection (WMMA)
    out_gemm_kernel<<<dim3(4, 256), 256, 0, stream>>>(Oh, Woh, bo, out);
}
